// WeightAndSum_macro_ring_35261681500385
// MI455X (gfx1250) — matrix operand order repaired and checked by reading, not compiled, not run on hardware
//
#include <hip/hip_runtime.h>
#include <stdint.h>

// ---------------------------------------------------------------------------
// WeightAndSum for MI455X (gfx1250): TDM-streamed tiles -> LDS, f32 VALU gate
// logits, WMMA (f16 A/B, f32 acc) weighted ragged segment-reduction, f32
// global atomics only at segment granularity. HBM-bound: ~475 MB @ 23.3 TB/s.
// ---------------------------------------------------------------------------

typedef __attribute__((ext_vector_type(16))) _Float16 v16h_t;
typedef __attribute__((ext_vector_type(8)))  float    v8f_t;
typedef __attribute__((ext_vector_type(4)))  uint32_t v4u_t;
typedef __attribute__((ext_vector_type(8)))  int      v8i_t;
typedef __attribute__((ext_vector_type(4)))  int      v4i_t;

#define FDIM      128
#define TILE      256            // rows per LDS tile
#define RSTRIDE   132            // 128 + 4-dword pad: 16B-aligned rows (b128 LDS)
#define NTHREADS  256            // 8 wave32 per block

// dynamic-LDS layout (dword offsets; no static LDS so dynamic base == 0)
#define BUF_DW    (TILE * RSTRIDE)
#define W_OFF     (2 * BUF_DW)       // W vector                 [128]
#define SEG_OFF   (W_OFF + FDIM)     // segment ids              [TILE]
#define PKI_OFF   (SEG_OFF + TILE)   // (slot<<16)|f16(gate)     [TILE]
#define SMAP_OFF  (PKI_OFF + TILE)   // slot -> seg id           [8 groups * 32]
#define SMAX_OFF  (SMAP_OFF + 8*32)  // max slot per group       [8]
#define SMEM_DW   (SMAX_OFF + 8)
#define SMEM_BYTES ((size_t)SMEM_DW * 4)

#if defined(__has_builtin)
#if __has_builtin(__builtin_amdgcn_tensor_load_to_lds)
#define HAVE_TDM 1
#endif
#endif
#ifndef HAVE_TDM
#define HAVE_TDM 0
#endif

// Issue one TDM 2D tile load: TILE x 128 f32 rows, +4 dword LDS pad per row.
// 6-arg builtin form (upstream clang-23 / therock-10.0 headers):
//   (uint32x4 g0, int32x8 g1, int32x4 g2, int32x4 g3, int32x8 pad, i32 cpol)
__device__ __forceinline__ void tdm_issue(const float* gsrc, uint32_t lds_byte) {
#if HAVE_TDM
  uint64_t ga = (uint64_t)(uintptr_t)gsrc;
  v4u_t g0;
  g0[0] = 1u;                                            // count=1, user mode
  g0[1] = lds_byte;                                      // lds_addr
  g0[2] = (uint32_t)ga;                                  // global_addr[31:0]
  g0[3] = (uint32_t)((ga >> 32) & 0x01FFFFFFu) | (2u << 30); // addr[56:32]|type=2
  v8i_t g1;
  // data_size=4B | pad_enable | pad_interval=128dw (code 6) | pad_amount=4dw (code 3)
  g1[0] = (2 << 16) | (1 << 20) | (6 << 22) | (3 << 25);
  g1[1] = (FDIM & 0xFFFF) << 16;              // tensor_dim0 = 128
  g1[2] = (TILE & 0xFFFF) << 16;              // tensor_dim1 = TILE
  g1[3] = (FDIM & 0xFFFF) << 16;              // tile_dim0 = 128
  g1[4] = TILE;                               // tile_dim1 = TILE
  g1[5] = FDIM;                               // tensor_dim0_stride = 128
  g1[6] = (FDIM & 0xFFFF) << 16;              // tensor_dim1_stride lo = 128
  g1[7] = 0;
  v4i_t z4 = (v4i_t)0;                        // 2D tensor: groups 2/3 unused
  v8i_t z8 = (v8i_t)0;
  __builtin_amdgcn_tensor_load_to_lds(g0, g1, z4, z4, z8, 0);
#else
  (void)gsrc; (void)lds_byte;
#endif
}

#if !HAVE_TDM
__device__ __forceinline__ void coop_load(float* dst, const float* src, int tid) {
  const float4* s4 = (const float4*)src;
  for (int idx = tid; idx < TILE * (FDIM / 4); idx += NTHREADS) {
    int r = idx >> 5, c = idx & 31;
    float4 v = s4[idx];
    *(float4*)(dst + r * RSTRIDE + c * 4) = v;
  }
}
#endif

__global__ void __launch_bounds__(NTHREADS)
weight_sum_wmma_kernel(const float* __restrict__ feats,
                       const float* __restrict__ W,
                       const float* __restrict__ bptr,
                       const int*   __restrict__ seg,
                       float*       __restrict__ out_sum,     // [G,128] accum
                       float*       __restrict__ out_logits,  // nullable
                       int total_tiles, int tiles_per_block) {
  extern __shared__ float smem[];
  float*    buf0 = smem;
  float*    buf1 = smem + BUF_DW;
  float*    Wl   = smem + W_OFF;
  int*      segl = (int*)(smem + SEG_OFF);
  uint32_t* pki  = (uint32_t*)(smem + PKI_OFF);
  int*      smap = (int*)(smem + SMAP_OFF);
  int*      smax = (int*)(smem + SMAX_OFF);

  const int tid  = threadIdx.x;
  const int lane = tid & 31;
  const int wv   = tid >> 5;          // wave id 0..7 == feature chunk
  const int h    = lane >> 4;         // lane-half (K split)
  const int n    = lane & 15;         // N column of B/D
  const int m    = lane & 15;         // M row of A

  if (tid < FDIM) Wl[tid] = W[tid];
  const float bias = bptr[0];
  const int tile0 = blockIdx.x * tiles_per_block;

#if HAVE_TDM
  if (tid < 32 && tile0 < total_tiles)
    tdm_issue(feats + (size_t)tile0 * TILE * FDIM, 0u);
#endif

  for (int t = 0; t < tiles_per_block; ++t) {
    const int tile = tile0 + t;
    if (tile >= total_tiles) break;
    float* buf = (t & 1) ? buf1 : buf0;

#if HAVE_TDM
    const bool more = (t + 1 < tiles_per_block) && (tile + 1 < total_tiles);
    if (tid < 32) {
      if (more)  // prefetch next tile into the other buffer
        tdm_issue(feats + (size_t)(tile + 1) * TILE * FDIM,
                  (uint32_t)(((t + 1) & 1) ? BUF_DW * 4 : 0));
      if (more) __builtin_amdgcn_s_wait_tensorcnt(1);  // current tile done
      else      __builtin_amdgcn_s_wait_tensorcnt(0);
    }
#else
    coop_load(buf, feats + (size_t)tile * TILE * FDIM, tid);
#endif
    __syncthreads();

    // ---- phase 1: f32 gate logit per row (thread == row), b128 LDS reads ----
    const int row_g = tile * TILE + tid;
    const float4* r4 = (const float4*)(buf + tid * RSTRIDE);
    const float4* w4 = (const float4*)Wl;
    float acc = bias;
#pragma unroll
    for (int j = 0; j < FDIM / 4; ++j) {
      float4 a = r4[j], b = w4[j];
      acc += a.x * b.x + a.y * b.y + a.z * b.z + a.w * b.w;
    }
    if (out_logits) out_logits[row_g] = acc;
    const float gate = 1.0f / (1.0f + __expf(-acc));
    segl[tid] = seg[row_g];
    __syncthreads();

    // ---- phase 1.5: per-32-row-group distinct-segment rank (wave scan);
    // pack (slot, f16 gate) into one dword for the A-matrix build ----
    {
      const int myseg = segl[tid];
      const int prev  = (lane == 0) ? myseg : segl[tid - 1];
      int sc = (lane != 0 && myseg != prev) ? 1 : 0;
      const int flag = sc;
      for (int d = 1; d < 32; d <<= 1) {
        int o = __shfl_up(sc, d, 32);
        if (lane >= d) sc += o;
      }
      const unsigned short gbits =
          __builtin_bit_cast(unsigned short, (_Float16)gate);
      pki[tid] = ((uint32_t)sc << 16) | (uint32_t)gbits;
      if (lane == 0 || flag) smap[wv * 32 + sc] = myseg;  // slot -> seg id
      if (lane == 31) smax[wv] = sc;
    }
    __syncthreads();

    // ---- phase 2: WMMA weighted segment-sum. wave wv owns features
    // [wv*16, wv*16+16); loops the 8 groups of 32 rows. ----
    const int f0 = wv * 16;
    const uint32_t lane_off = (uint32_t)(f0 + n);   // column offset in out row
#pragma unroll 1
    for (int gg = 0; gg < 8; ++gg) {
      const int base = gg * 32;
      const int nmax = smax[gg];
      // B[32x16] f16: element e of lane holds B[K=16h+e][n]
      v16h_t B;
      const float* gb = buf + base * RSTRIDE + f0 + n;
#pragma unroll
      for (int e = 0; e < 16; ++e)
        B[e] = (_Float16)gb[(h * 16 + e) * RSTRIDE];
      // unconditionally preload this lane's 16 packed (slot|gate) words
      uint32_t pk[16];
#pragma unroll
      for (int e = 0; e < 16; ++e) {
        const int k = 16 * h + e;  // K_A = 16h+e, matches B
        pk[e] = pki[base + k];
      }
      // up to 2 passes cover slots 0..31
      for (int pass = 0; pass <= (nmax >> 4); ++pass) {
        const uint32_t want = (uint32_t)(m + pass * 16);
        // A[16x32] f16 scatter matrix: branchless selects on registers
        v16h_t A;
#pragma unroll
        for (int e = 0; e < 16; ++e) {
          const _Float16 wh =
              __builtin_bit_cast(_Float16, (unsigned short)(pk[e] & 0xFFFFu));
          A[e] = ((pk[e] >> 16) == want) ? wh : (_Float16)0.0f;
        }
        v8f_t C = {};
        C = __builtin_amdgcn_wmma_f32_16x16x32_f16(
            false, A, false, B, (short)0, C, false, false);
        // batch-load slot->graph ids unconditionally (masked: always in-bounds)
        uint32_t gm[8];
#pragma unroll
        for (int v = 0; v < 8; ++v) {
          const int srow = (v + 8 * h + pass * 16) & 31;
          gm[v] = (uint32_t)smap[gg * 32 + srow];
        }
        // D rows: M = v + 8h; only the atomic itself is predicated.
        // 32-bit index from uniform base -> saddr atomic form.
#pragma unroll
        for (int v = 0; v < 8; ++v) {
          const int srow = v + 8 * h + pass * 16;
          const uint32_t idx = gm[v] * (uint32_t)FDIM + lane_off;
          if (srow <= nmax) {
            unsafeAtomicAdd(out_sum + idx, C[v]);
          }
        }
      }
    }
    __syncthreads();  // buffer free before it is re-targeted by TDM
  }
}

extern "C" void kernel_launch(void* const* d_in, const int* in_sizes, int n_in,
                              void* d_out, int out_size, void* d_ws, size_t ws_size,
                              hipStream_t stream) {
  (void)n_in; (void)d_ws; (void)ws_size;
  const float* atom_feats = (const float*)d_in[0];
  const float* vir_feats  = (const float*)d_in[1];
  const float* W_atom     = (const float*)d_in[2];
  const float* b_atom     = (const float*)d_in[3];
  const float* W_vir      = (const float*)d_in[4];
  const float* b_vir      = (const float*)d_in[5];
  const int*   atom_seg   = (const int*)d_in[6];
  const int*   vir_seg    = (const int*)d_in[7];

  const int n_atom   = in_sizes[0] / FDIM;   // 819200
  const int n_vir    = in_sizes[1] / FDIM;   // 65536
  const int n_graphs = (out_size - n_atom) / (2 * FDIM);  // 16384

  float* out        = (float*)d_out;
  float* out_atom   = out;
  float* out_vir    = out + (size_t)n_graphs * FDIM;
  float* out_logits = out + (size_t)2 * n_graphs * FDIM;

  // zero the accumulated regions (harness poisons buffers with 0xAA)
  (void)hipMemsetAsync(d_out, 0,
                       (size_t)2 * n_graphs * FDIM * sizeof(float), stream);

  (void)hipFuncSetAttribute((const void*)weight_sum_wmma_kernel,
                            hipFuncAttributeMaxDynamicSharedMemorySize,
                            (int)SMEM_BYTES);

  const int tpb = 2;  // two tiles per block -> TDM double-buffer depth 2
  const int tiles_a  = (n_atom + TILE - 1) / TILE;
  const int blocks_a = (tiles_a + tpb - 1) / tpb;
  weight_sum_wmma_kernel<<<blocks_a, NTHREADS, SMEM_BYTES, stream>>>(
      atom_feats, W_atom, b_atom, atom_seg, out_atom, out_logits, tiles_a, tpb);

  const int tiles_v  = (n_vir + TILE - 1) / TILE;
  const int blocks_v = (tiles_v + tpb - 1) / tpb;
  weight_sum_wmma_kernel<<<blocks_v, NTHREADS, SMEM_BYTES, stream>>>(
      vir_feats, W_vir, b_vir, vir_seg, out_vir, nullptr, tiles_v, tpb);
}
